// SubjectLayer_28887950033230
// MI455X (gfx1250) — matrix operand order repaired and checked by reading, not compiled, not run on hardware
//
#include <hip/hip_runtime.h>

#define B_     32
#define S_     2048
#define DIN    512
#define DOUT   512
#define NSUBJ  32

#define BM     128
#define BN     128
#define BK     32
#define PAD    8              // row stride = 40 bf16 = 80 bytes (16B aligned, conflict-free)
#define KROW   (BK + PAD)
#define THREADS 512           // 16 waves (wave32)

typedef __attribute__((ext_vector_type(16))) __bf16 v16bf;
typedef __attribute__((ext_vector_type(8)))  float  v8f;

union Frag {
    uint4 q[2];   // two ds_load_b128 worth of data
    v16bf v;      // 16 bf16 per lane = one WMMA A/B operand slice
};

__device__ __forceinline__ unsigned short f32_to_bf16_rne(float x) {
    unsigned u = __builtin_bit_cast(unsigned, x);
    unsigned r = u + 0x7FFFu + ((u >> 16) & 1u);   // round-to-nearest-even
    return (unsigned short)(r >> 16);
}
__device__ __forceinline__ float bf16_bits_to_f32(unsigned short h) {
    unsigned u = ((unsigned)h) << 16;
    return __builtin_bit_cast(float, u);
}
__device__ __forceinline__ void split_bf16(float x, unsigned short& hi, unsigned short& lo) {
    hi = f32_to_bf16_rne(x);
    float hf = bf16_bits_to_f32(hi);
    lo = f32_to_bf16_rne(x - hf);
}

__global__ __launch_bounds__(THREADS)
void subject_gemm_bf16x3(const float* __restrict__ X,
                         const float* __restrict__ subj,
                         const float* __restrict__ W,
                         float* __restrict__ Z)
{
    __shared__ unsigned short Ahi[BM][KROW];
    __shared__ unsigned short Alo[BM][KROW];
    __shared__ unsigned short Bhi[BN][KROW];   // stored transposed: [n][k]
    __shared__ unsigned short Blo[BN][KROW];
    __shared__ int s_subj;

    const int b    = blockIdx.z;
    const int m0   = blockIdx.x * BM;
    const int n0   = blockIdx.y * BN;
    const int tid  = threadIdx.x;
    const int lane = tid & 31;
    const int wave = tid >> 5;
    const int waveM = wave & 3;      // 4 waves along M (32 rows each)
    const int waveN = wave >> 2;     // 4 waves along N (32 cols each)
    const int laneM = lane & 15;
    const int kh    = lane >> 4;     // half-wave selector per ISA fragment layout

    // Recover subject index from the one-hot row (exact: one-hot entry is 1.0f)
    if (tid == 0) {
        int s = 0;
        for (int j = 0; j < NSUBJ; ++j)
            if (subj[b * NSUBJ + j] > 0.5f) s = j;
        s_subj = s;
    }
    __syncthreads();
    const int sidx = s_subj;

    v8f acc[2][2];
    #pragma unroll
    for (int mt = 0; mt < 2; ++mt)
        #pragma unroll
        for (int nt = 0; nt < 2; ++nt)
            acc[mt][nt] = (v8f){0.f, 0.f, 0.f, 0.f, 0.f, 0.f, 0.f, 0.f};

    const float* Xb = X + (size_t)b * S_ * DIN;
    const int mrow0 = waveM * 32;
    const int ncol0 = waveN * 32;

    auto gloadX = [&](int k0, float4 (&out)[2]) {
        #pragma unroll
        for (int i = 0; i < 2; ++i) {
            int f  = tid + THREADS * i;          // 0..1023 float4 slots
            int m  = f >> 3;
            int kq = (f & 7) << 2;
            out[i] = *(const float4*)(Xb + (size_t)(m0 + m) * DIN + k0 + kq);
        }
    };
    auto gloadW = [&](int k0, float (&out)[8]) {
        #pragma unroll
        for (int i = 0; i < 8; ++i) {
            int e  = tid + THREADS * i;          // 0..4095
            int kk = e >> 7;                     // 0..31
            int n  = e & 127;
            out[i] = W[((size_t)(k0 + kk) * DOUT + (n0 + n)) * NSUBJ + sidx];
        }
    };

    float4 xcur[2], xnxt[2];
    float  wcur[8], wnxt[8];
    gloadX(0, xcur);
    gloadW(0, wcur);

    for (int k0 = 0; k0 < DIN; k0 += BK) {
        if (k0) __syncthreads();                 // previous tile fully consumed

        // ---- issue next tile's global loads first: latency overlaps convert+WMMA
        if (k0 + BK < DIN) {
            gloadX(k0 + BK, xnxt);
            gloadW(k0 + BK, wnxt);
        }

        // ---- convert current X regs -> LDS bf16 hi/lo, row-major [m][k]
        #pragma unroll
        for (int i = 0; i < 2; ++i) {
            int f  = tid + THREADS * i;
            int m  = f >> 3;
            int kq = (f & 7) << 2;
            float xs[4] = {xcur[i].x, xcur[i].y, xcur[i].z, xcur[i].w};
            unsigned short h[4], l[4];
            #pragma unroll
            for (int j = 0; j < 4; ++j) split_bf16(xs[j], h[j], l[j]);
            uint2 hp, lp;
            hp.x = (unsigned)h[0] | ((unsigned)h[1] << 16);
            hp.y = (unsigned)h[2] | ((unsigned)h[3] << 16);
            lp.x = (unsigned)l[0] | ((unsigned)l[1] << 16);
            lp.y = (unsigned)l[2] | ((unsigned)l[3] << 16);
            *(uint2*)&Ahi[m][kq] = hp;
            *(uint2*)&Alo[m][kq] = lp;
        }
        // ---- convert current W_s regs -> LDS, stored transposed [n][k]
        #pragma unroll
        for (int i = 0; i < 8; ++i) {
            int e  = tid + THREADS * i;
            int kk = e >> 7;
            int n  = e & 127;
            unsigned short h, l;
            split_bf16(wcur[i], h, l);
            Bhi[n][kk] = h;
            Blo[n][kk] = l;
        }
        __syncthreads();

        // ---- fragments per ISA 16-bit A/B VGPR striping (two b128 loads each)
        Frag ah[2], al[2], bh[2], bl[2];
        #pragma unroll
        for (int mt = 0; mt < 2; ++mt) {
            int m = mrow0 + mt * 16 + laneM;
            ah[mt].q[0] = *(const uint4*)&Ahi[m][8 * kh];
            ah[mt].q[1] = *(const uint4*)&Ahi[m][16 + 8 * kh];
            al[mt].q[0] = *(const uint4*)&Alo[m][8 * kh];
            al[mt].q[1] = *(const uint4*)&Alo[m][16 + 8 * kh];
        }
        #pragma unroll
        for (int nt = 0; nt < 2; ++nt) {
            int n = ncol0 + nt * 16 + laneM;
            bh[nt].q[0] = *(const uint4*)&Bhi[n][8 * kh];
            bh[nt].q[1] = *(const uint4*)&Bhi[n][16 + 8 * kh];
            bl[nt].q[0] = *(const uint4*)&Blo[n][8 * kh];
            bl[nt].q[1] = *(const uint4*)&Blo[n][16 + 8 * kh];
        }

        // ---- bf16x3: hi*hi + hi*lo + lo*hi, f32 accumulate
        #pragma unroll
        for (int mt = 0; mt < 2; ++mt)
            #pragma unroll
            for (int nt = 0; nt < 2; ++nt) {
                acc[mt][nt] = __builtin_amdgcn_wmma_f32_16x16x32_bf16(
                    false, ah[mt].v, false, bh[nt].v, (short)0, acc[mt][nt], false, false);
                acc[mt][nt] = __builtin_amdgcn_wmma_f32_16x16x32_bf16(
                    false, ah[mt].v, false, bl[nt].v, (short)0, acc[mt][nt], false, false);
                acc[mt][nt] = __builtin_amdgcn_wmma_f32_16x16x32_bf16(
                    false, al[mt].v, false, bh[nt].v, (short)0, acc[mt][nt], false, false);
            }

        // ---- rotate register staging buffers
        #pragma unroll
        for (int i = 0; i < 2; ++i) xcur[i] = xnxt[i];
        #pragma unroll
        for (int i = 0; i < 8; ++i) wcur[i] = wnxt[i];
    }

    // ---- epilogue: C/D layout = VGPR p holds (M = p + 8*kh, N = laneM)
    // Z is write-once: use non-temporal stores so it doesn't evict the
    // reused X/W working set (~168 MB) from the 192 MB L2.
    float* Zb = Z + (size_t)b * S_ * DOUT;
    #pragma unroll
    for (int mt = 0; mt < 2; ++mt)
        #pragma unroll
        for (int nt = 0; nt < 2; ++nt) {
            int gm = m0 + mrow0 + mt * 16 + 8 * kh;
            int gn = n0 + ncol0 + nt * 16 + laneM;
            #pragma unroll
            for (int p = 0; p < 8; ++p)
                __builtin_nontemporal_store(acc[mt][nt][p],
                                            &Zb[(size_t)(gm + p) * DOUT + gn]);
        }
}

extern "C" void kernel_launch(void* const* d_in, const int* in_sizes, int n_in,
                              void* d_out, int out_size, void* d_ws, size_t ws_size,
                              hipStream_t stream) {
    const float* X    = (const float*)d_in[0];
    const float* subj = (const float*)d_in[1];
    const float* W    = (const float*)d_in[2];
    float* Z          = (float*)d_out;

    dim3 grid(S_ / BM, DOUT / BN, B_);   // 16 x 4 x 32 = 2048 blocks
    subject_gemm_bf16x3<<<grid, THREADS, 0, stream>>>(X, subj, W, Z);
}
